// MultiHeadAttentionNoBatch_40080634806696
// MI455X (gfx1250) — matrix operand order repaired and checked by reading, not compiled, not run on hardware
//
#include <hip/hip_runtime.h>

// Problem constants (match reference)
constexpr int kT     = 2048;
constexpr int kDin   = 1024;
constexpr int kDout  = 1024;
constexpr int kHeads = 16;
constexpr int kHd    = 64;   // head dim; reference quirk: scores scaled by sqrt(64)=8

typedef __attribute__((ext_vector_type(16))) __bf16 v16bf;
typedef __attribute__((ext_vector_type(8)))  __bf16 v8bf;
typedef __attribute__((ext_vector_type(8)))  float  v8f;

__device__ __forceinline__ __bf16 bits_to_bf16(unsigned short u) {
    union { unsigned short s; __bf16 h; } c; c.s = u; return c.h;
}

// Load a contiguous bf16 pair (4B-aligned) into fragment slots 2i, 2i+1.
__device__ __forceinline__ void ld_pair(const __bf16* p, v16bf& f, int i) {
    const unsigned int u = *reinterpret_cast<const unsigned int*>(p);
    f[2 * i]     = bits_to_bf16((unsigned short)(u & 0xffffu));
    f[2 * i + 1] = bits_to_bf16((unsigned short)(u >> 16));
}

// A-fragment K offset pattern for 16-bit 16x32 A matrices (ISA 7.12.2):
// VGPR i, lane-half hi: K = (i<4?0:16) + hi*8 + (i%4)*2  (pair K, K+1)
__device__ __forceinline__ int kpat(int i, int hi) {
    return ((i & 4) << 2) + hi * 8 + ((i & 3) << 1);
}

// ---------------------------------------------------------------------------
// Kernel 0a: x (f32) -> xb (bf16), vectorized. 8 elements/thread.
// ---------------------------------------------------------------------------
__global__ __launch_bounds__(256)
void cvt_x(const float* __restrict__ x, __bf16* __restrict__ xb) {
    const size_t i = ((size_t)blockIdx.x * 256 + threadIdx.x) * 8;
    const float4 f0 = *reinterpret_cast<const float4*>(x + i);
    const float4 f1 = *reinterpret_cast<const float4*>(x + i + 4);
    v8bf o;
    o[0] = (__bf16)f0.x; o[1] = (__bf16)f0.y; o[2] = (__bf16)f0.z; o[3] = (__bf16)f0.w;
    o[4] = (__bf16)f1.x; o[5] = (__bf16)f1.y; o[6] = (__bf16)f1.z; o[7] = (__bf16)f1.w;
    *reinterpret_cast<v8bf*>(xb + i) = o;
}

// ---------------------------------------------------------------------------
// Kernel 0b: Wt = W^T as bf16.  W is [Dout][Din] f32; Wt is [Din][Dout] bf16.
// LDS-tiled 32x32 transpose; blockIdx.z selects {Wq,Wk,Wv}.
// ---------------------------------------------------------------------------
__global__ __launch_bounds__(256)
void transpose_w(const float* __restrict__ Wq, const float* __restrict__ Wk,
                 const float* __restrict__ Wv, __bf16* __restrict__ Wt3) {
    __shared__ __bf16 tile[32][33];
    const int z = blockIdx.z;
    const float* W = (z == 0) ? Wq : (z == 1) ? Wk : Wv;
    __bf16* Wt = Wt3 + (size_t)z * kDin * kDout;

    const int kBase = blockIdx.x * 32;   // Din direction
    const int nBase = blockIdx.y * 32;   // Dout direction
    const int tx = threadIdx.x & 31;
    const int ty = threadIdx.x >> 5;     // 0..7

#pragma unroll
    for (int q = 0; q < 4; ++q) {
        const int n = ty * 4 + q;
        tile[n][tx] = (__bf16)W[(size_t)(nBase + n) * kDin + kBase + tx];
    }
    __syncthreads();
#pragma unroll
    for (int q = 0; q < 4; ++q) {
        const int k = ty * 4 + q;
        Wt[(size_t)(kBase + k) * kDout + nBase + tx] = tile[tx][k];
    }
}

// ---------------------------------------------------------------------------
// Kernel 1: Y = xb @ Wt  (bf16 WMMA, f32 accumulate).
// Block = 256 thr = 8 waves arranged 4(M) x 2(N); block tile 128 x 128.
// Wave tile: 32(M) x 64(N) -> 8 WMMAs / K-step fed by 16 packed b32 A loads + 64 bf16 B loads.
// z==1 (K) is stored transposed: Kt[Dout][T].
// ---------------------------------------------------------------------------
__global__ __launch_bounds__(256)
void qkv_gemm(const __bf16* __restrict__ xb, const __bf16* __restrict__ Wt3,
              __bf16* __restrict__ Qw, __bf16* __restrict__ Kt,
              __bf16* __restrict__ Vw) {
    const int lane = threadIdx.x & 31;
    const int wave = threadIdx.x >> 5;
    const int hi   = lane >> 4;
    const int l16  = lane & 15;
    const int z    = blockIdx.z;

    const __bf16* Wt = Wt3 + (size_t)z * kDin * kDout;

    const int mw = wave & 3;           // 0..3
    const int nw = wave >> 2;          // 0..1
    const int rowBase = blockIdx.x * 128 + mw * 32;
    const int colBase = blockIdx.y * 128 + nw * 64;

    v8f acc[2][4] = {};

    for (int k0 = 0; k0 < kDin; k0 += 32) {
        // Two A fragments (rows rowBase..+15, rowBase+16..+31): packed pairs.
        v16bf a[2];
#pragma unroll
        for (int t = 0; t < 2; ++t) {
            const __bf16* xrow = xb + (size_t)(rowBase + t * 16 + l16) * kDin + k0;
#pragma unroll
            for (int i = 0; i < 8; ++i) ld_pair(xrow + kpat(i, hi), a[t], i);
        }
        // Four B fragments: B[k][n] = Wt[k0+k][colBase+n]; lane = column n,
        // VGPR i holds k = kpat(i,hi), kpat(i,hi)+1 (strided in Wt).
#pragma unroll
        for (int j = 0; j < 4; ++j) {
            v16bf b;
            const __bf16* wrow =
                Wt + (size_t)k0 * kDout + colBase + j * 16 + l16;
#pragma unroll
            for (int i = 0; i < 8; ++i) {
                b[2 * i]     = wrow[(size_t)kpat(i, hi) * kDout];
                b[2 * i + 1] = wrow[(size_t)(kpat(i, hi) + 1) * kDout];
            }
#pragma unroll
            for (int t = 0; t < 2; ++t)
                acc[t][j] = __builtin_amdgcn_wmma_f32_16x16x32_bf16(
                    false, a[t], false, b, (short)0, acc[t][j], false, false);
        }
    }

    // Store. D layout: row = base + r + 8*hi, col = colBase + j*16 + l16.
    if (z == 1) {
        // K stored transposed: Kt[col][row]
#pragma unroll
        for (int t = 0; t < 2; ++t)
#pragma unroll
            for (int j = 0; j < 4; ++j)
#pragma unroll
                for (int r = 0; r < 8; ++r) {
                    const int row = rowBase + t * 16 + r + 8 * hi;
                    const int col = colBase + j * 16 + l16;
                    Kt[(size_t)col * kT + row] = (__bf16)acc[t][j][r];
                }
    } else {
        __bf16* Y = (z == 0) ? Qw : Vw;
#pragma unroll
        for (int t = 0; t < 2; ++t)
#pragma unroll
            for (int j = 0; j < 4; ++j)
#pragma unroll
                for (int r = 0; r < 8; ++r) {
                    const int row = rowBase + t * 16 + r + 8 * hi;
                    const int col = colBase + j * 16 + l16;
                    Y[(size_t)row * kDout + col] = (__bf16)acc[t][j][r];
                }
    }
}

// ---------------------------------------------------------------------------
// Kernel 2: causal flash attention, one wave per (head, 16-query tile).
// Block = 128 thr = 4 waves; grid = (qTiles/4, heads).
// ---------------------------------------------------------------------------
__global__ __launch_bounds__(128)
void flash_attn(const __bf16* __restrict__ Q,   // [T][Dout] bf16
                const __bf16* __restrict__ Kt,  // [Dout][T] bf16 (transposed)
                const __bf16* __restrict__ V,   // [T][Dout] bf16
                float* __restrict__ out) {
    __shared__ __bf16 plds[4][16][36];  // per-wave P staging, padded rows (72B)

    const int lane = threadIdx.x & 31;
    const int wave = threadIdx.x >> 5;
    const int hi   = lane >> 4;
    const int l16  = lane & 15;
    const int head = blockIdx.y;
    const int qt   = blockIdx.x * 4 + wave;
    const int qBase = qt * 16;
    const int hd    = head * kHd;

    // Q fragments (A-matrix 16x32, two dim-chunks covering Dh=64)
    v16bf qf[2];
    {
        const __bf16* qrow = Q + (size_t)(qBase + l16) * kDout + hd;
#pragma unroll
        for (int kk = 0; kk < 2; ++kk)
#pragma unroll
            for (int i = 0; i < 8; ++i)
                ld_pair(qrow + kk * 32 + kpat(i, hi), qf[kk], i);
    }

    float m[8], l[8];
    v8f o[4] = {};
#pragma unroll
    for (int r = 0; r < 8; ++r) { m[r] = -3.0e38f; l[r] = 0.0f; }

    const int nkb = (qBase + 15) / 32 + 1;  // causal key-block count (blocks of 32)
    for (int kb = 0; kb < nkb; ++kb) {
        const int keyBase = kb * 32;

        // ---- scores: two 16x16 tiles (keys keyBase..+15, +16..+31) ----
        v8f s[2] = {};
#pragma unroll
        for (int j = 0; j < 2; ++j) {
#pragma unroll
            for (int kk = 0; kk < 2; ++kk) {
                // B[k=dim][n=key] = Kt[hd + dim][key]; lane = key column,
                // elements = dims kpat(i,hi), +1 (strided in Kt).
                v16bf b;
                const __bf16* kp =
                    Kt + (size_t)(hd + kk * 32) * kT + keyBase + j * 16 + l16;
#pragma unroll
                for (int i = 0; i < 8; ++i) {
                    b[2 * i]     = kp[(size_t)kpat(i, hi) * kT];
                    b[2 * i + 1] = kp[(size_t)(kpat(i, hi) + 1) * kT];
                }
                s[j] = __builtin_amdgcn_wmma_f32_16x16x32_bf16(
                    false, qf[kk], false, b, (short)0, s[j], false, false);
            }
        }

        // ---- scale x8 (reference quirk), causal mask, online softmax ----
        float p0[8], p1[8];
#pragma unroll
        for (int r = 0; r < 8; ++r) {
            const int row = qBase + r + 8 * hi;
            const int c0  = keyBase + l16;
            const int c1  = keyBase + 16 + l16;
            float s0 = (c0 <= row) ? s[0][r] * 8.0f : -3.0e38f;
            float s1 = (c1 <= row) ? s[1][r] * 8.0f : -3.0e38f;
            float mx = fmaxf(s0, s1);
#pragma unroll
            for (int off = 1; off < 16; off <<= 1)
                mx = fmaxf(mx, __shfl_xor(mx, off, 16));
            const float mn    = fmaxf(m[r], mx);
            const float alpha = __expf(m[r] - mn);
            const float e0 = __expf(s0 - mn);
            const float e1 = __expf(s1 - mn);
            float rs = e0 + e1;
#pragma unroll
            for (int off = 1; off < 16; off <<= 1)
                rs += __shfl_xor(rs, off, 16);
            l[r] = l[r] * alpha + rs;
            m[r] = mn;
            p0[r] = e0;
            p1[r] = e1;
#pragma unroll
            for (int j = 0; j < 4; ++j) o[j][r] *= alpha;
        }

        // ---- relayout P: C/D tile -> LDS -> A fragment ----
#pragma unroll
        for (int r = 0; r < 8; ++r) {
            plds[wave][r + 8 * hi][l16]      = (__bf16)p0[r];
            plds[wave][r + 8 * hi][16 + l16] = (__bf16)p1[r];
        }
        __builtin_amdgcn_wave_barrier();
        asm volatile("s_wait_dscnt 0x0" ::: "memory");  // CDNA5 split DS counter
        v16bf pf;
#pragma unroll
        for (int i = 0; i < 8; ++i) {
            const int kk = kpat(i, hi);
            const unsigned int u =
                *reinterpret_cast<const unsigned int*>(&plds[wave][l16][kk]);
            pf[2 * i]     = bits_to_bf16((unsigned short)(u & 0xffffu));
            pf[2 * i + 1] = bits_to_bf16((unsigned short)(u >> 16));
        }
        __builtin_amdgcn_wave_barrier();

        // ---- O += P x V : B[k=key][n=d]; lane = d column, elements = keys kpat(i,hi), +1 ----
#pragma unroll
        for (int j = 0; j < 4; ++j) {
            v16bf b;
            const __bf16* vp = V + (size_t)keyBase * kDout + hd + j * 16 + l16;
#pragma unroll
            for (int i = 0; i < 8; ++i) {
                b[2 * i]     = vp[(size_t)kpat(i, hi) * kDout];
                b[2 * i + 1] = vp[(size_t)(kpat(i, hi) + 1) * kDout];
            }
            o[j] = __builtin_amdgcn_wmma_f32_16x16x32_bf16(
                false, pf, false, b, (short)0, o[j], false, false);
        }
    }

    // ---- epilogue: divide by softmax denominator, store f32 ----
#pragma unroll
    for (int j = 0; j < 4; ++j)
#pragma unroll
        for (int r = 0; r < 8; ++r) {
            const int row = qBase + r + 8 * hi;
            out[(size_t)row * kDout + hd + j * 16 + l16] = o[j][r] / l[r];
        }
}

// ---------------------------------------------------------------------------
extern "C" void kernel_launch(void* const* d_in, const int* in_sizes, int n_in,
                              void* d_out, int out_size, void* d_ws, size_t ws_size,
                              hipStream_t stream) {
    const float* x  = (const float*)d_in[0];
    const float* Wq = (const float*)d_in[1];
    const float* Wk = (const float*)d_in[2];
    const float* Wv = (const float*)d_in[3];

    // Workspace layout (bf16):
    //   xb  [T][Din]          4 MB
    //   Wt3 3x [Din][Dout]    6 MB
    //   Qw  [T][Dout]         4 MB
    //   Kt  [Dout][T]         4 MB
    //   Vw  [T][Dout]         4 MB
    __bf16* xb  = (__bf16*)d_ws;
    __bf16* Wt3 = xb  + (size_t)kT * kDin;
    __bf16* Qw  = Wt3 + (size_t)3 * kDin * kDout;
    __bf16* Kt  = Qw  + (size_t)kT * kDout;
    __bf16* Vw  = Kt  + (size_t)kDout * kT;

    // 0a: x -> bf16
    cvt_x<<<(kT * kDin) / (256 * 8), 256, 0, stream>>>(x, xb);
    // 0b: W -> W^T bf16 (3 matrices)
    transpose_w<<<dim3(kDin / 32, kDout / 32, 3), 256, 0, stream>>>(Wq, Wk, Wv, Wt3);
    // 1: QKV projections; grid (T/128, Dout/128, 3)
    qkv_gemm<<<dim3(kT / 128, kDout / 128, 3), 256, 0, stream>>>(xb, Wt3, Qw, Kt, Vw);
    // 2: attention: 128 query tiles, 4 waves/block -> grid (32, heads)
    flash_attn<<<dim3((kT / 16) / 4, kHeads), 128, 0, stream>>>(Qw, Kt, Vw, (float*)d_out);
}
